// jointBert_9861244911778
// MI455X (gfx1250) — matrix operand order repaired and checked by reading, not compiled, not run on hardware
//
#include <hip/hip_runtime.h>
#include <hip/hip_bf16.h>

// ---------------- problem constants ----------------
#define Bc   32
#define Sc   256
#define Dc   768
#define Lc   6
#define Hc   12
#define FFc  3072
#define Kt   128      // CRF tag count
#define NIc  64
#define DHc  64

typedef __attribute__((ext_vector_type(16))) __bf16 bf16x16;
typedef __attribute__((ext_vector_type(8)))  float  floatx8;

union BfFrag { bf16x16 v; unsigned int u[8]; };

__device__ __forceinline__ unsigned short f2bf(float f) {
  unsigned int u = __float_as_uint(f);
  u += 0x7FFFu + ((u >> 16) & 1u);   // round-to-nearest-even
  return (unsigned short)(u >> 16);
}
__device__ __forceinline__ unsigned int pack2bf(float a, float b) {
  return (unsigned int)f2bf(a) | ((unsigned int)f2bf(b) << 16);
}

// fragment gather + 2x2 WMMA accumulate (shared by fast/slow loops)
__device__ __forceinline__ void mma_step(const unsigned short (*As)[40],
                                         const unsigned short (*Bs)[40],
                                         int wm, int wn, int l15, int khalf,
                                         floatx8 acc[2][2]) {
  BfFrag af[2], bf[2];
  #pragma unroll
  for (int mi = 0; mi < 2; ++mi) {
    int m = wm + mi * 16 + l15;
    #pragma unroll
    for (int g = 0; g < 8; ++g) {
      int k = ((g & 3) * 2) + ((g >> 2) * 16) + khalf * 8;
      af[mi].u[g] = *(const unsigned int*)&As[m][k];
    }
  }
  #pragma unroll
  for (int ni = 0; ni < 2; ++ni) {
    int n = wn + ni * 16 + l15;
    #pragma unroll
    for (int g = 0; g < 8; ++g) {
      int k = ((g & 3) * 2) + ((g >> 2) * 16) + khalf * 8;
      bf[ni].u[g] = *(const unsigned int*)&Bs[n][k];
    }
  }
  #pragma unroll
  for (int mi = 0; mi < 2; ++mi)
    #pragma unroll
    for (int ni = 0; ni < 2; ++ni)
      acc[mi][ni] = __builtin_amdgcn_wmma_f32_16x16x32_bf16(
          false, af[mi].v, false, bf[ni].v, (short)0, acc[mi][ni], false, false);
}

// =====================================================================
// Generic batched GEMM:  C = alpha * (A @ B) + bias
//   A: fp32 [M,Kd] row-major, row stride lda, batch offset zo*sAo + zi*sAi
//   B: fp32, bT=0 -> [Kd,N] (ldb = row stride), bT=1 -> [N,Kd]
//   C: fp32 [M,N], row stride ldc
// bf16 WMMA 16x16x32, f32 accumulate. 256 threads, tile 128x64.
// Unswitched: tight vectorized loop for full tiles, guarded loop otherwise.
// =====================================================================
__global__ __launch_bounds__(256)
void wmma_gemm_bf16(const float* __restrict__ A, long lda, long sAo, long sAi,
                    const float* __restrict__ Bm, long ldb, long sBo, long sBi, int bT,
                    float* __restrict__ C, long ldc, long sCo, long sCi,
                    const float* __restrict__ bias,
                    int M, int N, int Kd, int innerB, float alpha) {
  __shared__ unsigned short As[128][40];   // [m][k], padded (80B row, 8B aligned)
  __shared__ unsigned short Bs[64][40];    // [n][k], padded

  int z  = blockIdx.z;
  int zo = z / innerB, zi = z - zo * innerB;
  const float* Ab = A  + (long)zo * sAo + (long)zi * sAi;
  const float* Bb = Bm + (long)zo * sBo + (long)zi * sBi;
  float*       Cb = C  + (long)zo * sCo + (long)zi * sCi;

  int tid  = threadIdx.x;
  int m0   = blockIdx.y * 128;
  int n0   = blockIdx.x * 64;
  int wave = tid >> 5, lane = tid & 31;
  int wm   = (wave & 3) * 32;   // wave row offset inside tile
  int wn   = (wave >> 2) * 32;  // wave col offset inside tile
  int l15  = lane & 15;
  int khalf = lane >> 4;

  const floatx8 vzero = {0.f,0.f,0.f,0.f,0.f,0.f,0.f,0.f};
  floatx8 acc[2][2];
  acc[0][0] = vzero; acc[0][1] = vzero; acc[1][0] = vzero; acc[1][1] = vzero;

  const bool fullTile = (m0 + 128 <= M) && (n0 + 64 <= N) && ((Kd & 31) == 0);

  if (fullTile) {
    // ------------- tight vectorized loop (all encoder GEMMs) -------------
    // precomputed per-thread staging coordinates
    int ar  = tid >> 3;             // A row for i=0 slot (of 4 slots, +32 rows each)
    int ac4 = (tid & 7) * 4;        // A col (float4 granularity)
    if (bT) {
      int bn  = tid >> 3;           // 0..31 (+32 for i=1)
      int bc4 = (tid & 7) * 4;
      for (int k0 = 0; k0 < Kd; k0 += 32) {
        #pragma unroll
        for (int i = 0; i < 4; ++i) {
          int r = ar + i * 32;
          const float4 vv = *(const float4*)(Ab + (long)(m0 + r) * lda + k0 + ac4);
          unsigned long long p =
              (unsigned long long)pack2bf(vv.x, vv.y) |
              ((unsigned long long)pack2bf(vv.z, vv.w) << 32);
          *(unsigned long long*)&As[r][ac4] = p;
        }
        #pragma unroll
        for (int i = 0; i < 2; ++i) {
          int n = bn + i * 32;
          const float4 vv = *(const float4*)(Bb + (long)(n0 + n) * ldb + k0 + bc4);
          unsigned long long p =
              (unsigned long long)pack2bf(vv.x, vv.y) |
              ((unsigned long long)pack2bf(vv.z, vv.w) << 32);
          *(unsigned long long*)&Bs[n][bc4] = p;
        }
        __syncthreads();
        mma_step(As, Bs, wm, wn, l15, khalf, acc);
        __syncthreads();
      }
    } else {
      int bk  = tid >> 4;           // 0..15 (+16 for i=1)
      int bn4 = (tid & 15) * 4;
      for (int k0 = 0; k0 < Kd; k0 += 32) {
        #pragma unroll
        for (int i = 0; i < 4; ++i) {
          int r = ar + i * 32;
          const float4 vv = *(const float4*)(Ab + (long)(m0 + r) * lda + k0 + ac4);
          unsigned long long p =
              (unsigned long long)pack2bf(vv.x, vv.y) |
              ((unsigned long long)pack2bf(vv.z, vv.w) << 32);
          *(unsigned long long*)&As[r][ac4] = p;
        }
        #pragma unroll
        for (int i = 0; i < 2; ++i) {
          int kk = bk + i * 16;
          const float4 vv = *(const float4*)(Bb + (long)(k0 + kk) * ldb + n0 + bn4);
          Bs[bn4 + 0][kk] = f2bf(vv.x);
          Bs[bn4 + 1][kk] = f2bf(vv.y);
          Bs[bn4 + 2][kk] = f2bf(vv.z);
          Bs[bn4 + 3][kk] = f2bf(vv.w);
        }
        __syncthreads();
        mma_step(As, Bs, wm, wn, l15, khalf, acc);
        __syncthreads();
      }
    }
  } else {
    // ------------- guarded scalar loop (intent head, ragged edges) -------------
    for (int k0 = 0; k0 < Kd; k0 += 32) {
      #pragma unroll
      for (int i = 0; i < 16; ++i) {
        int idx = tid + i * 256;
        int r = idx >> 5, c = idx & 31;
        int gm = m0 + r, gk = k0 + c;
        float vld = (gm < M && gk < Kd) ? Ab[(long)gm * lda + gk] : 0.f;
        As[r][c] = f2bf(vld);
      }
      if (bT) {
        #pragma unroll
        for (int i = 0; i < 8; ++i) {
          int idx = tid + i * 256;
          int n = idx >> 5, c = idx & 31;
          int gn = n0 + n, gk = k0 + c;
          float vld = (gn < N && gk < Kd) ? Bb[(long)gn * ldb + gk] : 0.f;
          Bs[n][c] = f2bf(vld);
        }
      } else {
        #pragma unroll
        for (int i = 0; i < 8; ++i) {
          int idx = tid + i * 256;
          int c = idx >> 6, n = idx & 63;
          int gn = n0 + n, gk = k0 + c;
          float vld = (gn < N && gk < Kd) ? Bb[(long)gk * ldb + gn] : 0.f;
          Bs[n][c] = f2bf(vld);
        }
      }
      __syncthreads();
      mma_step(As, Bs, wm, wn, l15, khalf, acc);
      __syncthreads();
    }
  }

  // ---- store: lane = (m>=8?16:0)+n, vgpr r = m&7 ----
  int mh = khalf * 8;
  #pragma unroll
  for (int mi = 0; mi < 2; ++mi)
    #pragma unroll
    for (int ni = 0; ni < 2; ++ni) {
      int n = n0 + wn + ni * 16 + l15;
      if (n >= N) continue;
      float bv = bias ? bias[n] : 0.f;
      #pragma unroll
      for (int r = 0; r < 8; ++r) {
        int m = m0 + wm + mi * 16 + mh + r;
        if (m < M) Cb[(long)m * ldc + n] = alpha * acc[mi][ni][r] + bv;
      }
    }
}

// =====================================================================
// Embedding + LayerNorm: x[b,s,:] = LN(word_emb[id] + pos_emb[s])
// =====================================================================
__global__ __launch_bounds__(256)
void embed_ln_kernel(const int* __restrict__ ids, const float* __restrict__ wemb,
                     const float* __restrict__ pemb, const float* __restrict__ sc,
                     const float* __restrict__ bi, float* __restrict__ x) {
  int tok = blockIdx.x;
  int ss  = tok & (Sc - 1);
  int id  = ids[tok];
  const float* w = wemb + (long)id * Dc;
  const float* p = pemb + (long)ss * Dc;
  __shared__ float r1[256], r2[256];
  float vals[3]; float s = 0.f, q = 0.f;
  #pragma unroll
  for (int i = 0; i < 3; ++i) {
    int d = threadIdx.x + i * 256;
    float v = w[d] + p[d];
    vals[i] = v; s += v; q += v * v;
  }
  r1[threadIdx.x] = s; r2[threadIdx.x] = q; __syncthreads();
  for (int off = 128; off > 0; off >>= 1) {
    if (threadIdx.x < off) { r1[threadIdx.x] += r1[threadIdx.x + off]; r2[threadIdx.x] += r2[threadIdx.x + off]; }
    __syncthreads();
  }
  float mean = r1[0] / (float)Dc;
  float var  = r2[0] / (float)Dc - mean * mean;
  float rstd = rsqrtf(var + 1e-12f);
  #pragma unroll
  for (int i = 0; i < 3; ++i) {
    int d = threadIdx.x + i * 256;
    x[(long)tok * Dc + d] = (vals[i] - mean) * rstd * sc[d] + bi[d];
  }
}

// x = LN(x + delta) in place, one block per token
__global__ __launch_bounds__(256)
void residual_ln_kernel(float* __restrict__ x, const float* __restrict__ delta,
                        const float* __restrict__ sc, const float* __restrict__ bi) {
  int tok = blockIdx.x;
  __shared__ float r1[256], r2[256];
  float vals[3]; float s = 0.f, q = 0.f;
  #pragma unroll
  for (int i = 0; i < 3; ++i) {
    int d = threadIdx.x + i * 256;
    float v = x[(long)tok * Dc + d] + delta[(long)tok * Dc + d];
    vals[i] = v; s += v; q += v * v;
  }
  r1[threadIdx.x] = s; r2[threadIdx.x] = q; __syncthreads();
  for (int off = 128; off > 0; off >>= 1) {
    if (threadIdx.x < off) { r1[threadIdx.x] += r1[threadIdx.x + off]; r2[threadIdx.x] += r2[threadIdx.x + off]; }
    __syncthreads();
  }
  float mean = r1[0] / (float)Dc;
  float var  = r2[0] / (float)Dc - mean * mean;
  float rstd = rsqrtf(var + 1e-12f);
  #pragma unroll
  for (int i = 0; i < 3; ++i) {
    int d = threadIdx.x + i * 256;
    x[(long)tok * Dc + d] = (vals[i] - mean) * rstd * sc[d] + bi[d];
  }
}

// exact GELU, one element per thread
__global__ __launch_bounds__(256)
void gelu_kernel(float* __restrict__ h, long n) {
  long i = (long)blockIdx.x * 256 + threadIdx.x;
  if (i < n) {
    float v = h[i];
    h[i] = 0.5f * v * (1.f + erff(v * 0.70710678118654752f));
  }
}

// masked softmax over last dim (S=256); one block per (b,h,q) row
__global__ __launch_bounds__(256)
void softmax_mask_kernel(float* __restrict__ scmat, const int* __restrict__ mask) {
  long row = blockIdx.x;
  int b = (int)(row / (Hc * Sc));
  float* r = scmat + row * Sc;
  int j = threadIdx.x;
  float v = r[j] + ((mask[b * Sc + j] > 0) ? 0.f : -1e9f);
  __shared__ float red[256];
  red[j] = v; __syncthreads();
  for (int off = 128; off > 0; off >>= 1) {
    if (j < off) red[j] = fmaxf(red[j], red[j + off]);
    __syncthreads();
  }
  float mx = red[0]; __syncthreads();
  float e = __expf(v - mx);
  red[j] = e; __syncthreads();
  for (int off = 128; off > 0; off >>= 1) {
    if (j < off) red[j] += red[j + off];
    __syncthreads();
  }
  r[j] = e / red[0];
}

// transpose 128x128 CRF transition matrix: trT[j*K+i] = trans[i*K+j]
__global__ __launch_bounds__(256)
void transpose128_kernel(const float* __restrict__ in, float* __restrict__ outT) {
  int i = blockIdx.x * 256 + threadIdx.x;
  if (i < Kt * Kt) {
    int r = i >> 7, c = i & 127;
    outT[c * Kt + r] = in[i];
  }
}

// ---------------- CRF forward (log-partition), single workgroup ----------------
__global__ __launch_bounds__(1024)
void crf_forward_kernel(const float* __restrict__ emis, const int* __restrict__ mask,
                        const float* __restrict__ start, const float* __restrict__ trT,
                        float* __restrict__ scoreT) {
  __shared__ float cur[Bc * Kt];
  __shared__ float nxt[Bc * Kt];
  int tid = threadIdx.x;
  for (int st = tid; st < Bc * Kt; st += 1024) {
    int bb = st >> 7, j = st & 127;
    cur[st] = start[j] + emis[((long)bb * Sc) * Kt + j];
  }
  __syncthreads();
  for (int t = 1; t < Sc; ++t) {
    for (int st = tid; st < Bc * Kt; st += 1024) {
      int bb = st >> 7, j = st & 127;
      const float* c = &cur[bb * Kt];
      const float* tr = &trT[j * Kt];
      float mx = -1e30f;
      for (int i = 0; i < Kt; ++i) mx = fmaxf(mx, c[i] + tr[i]);
      float ssum = 0.f;
      for (int i = 0; i < Kt; ++i) ssum += __expf(c[i] + tr[i] - mx);
      float nv = mx + __logf(ssum) + emis[((long)bb * Sc + t) * Kt + j];
      nxt[st] = (mask[bb * Sc + t] > 0) ? nv : c[j];
    }
    __syncthreads();
    for (int st = tid; st < Bc * Kt; st += 1024) cur[st] = nxt[st];
    __syncthreads();
  }
  for (int st = tid; st < Bc * Kt; st += 1024) scoreT[st] = cur[st];
}

// ---------------- Viterbi forward pass, single workgroup ----------------
__global__ __launch_bounds__(1024)
void crf_viterbi_kernel(const float* __restrict__ emis, const int* __restrict__ mask,
                        const float* __restrict__ start, const float* __restrict__ trT,
                        float* __restrict__ scoreV, int* __restrict__ bp) {
  __shared__ float cur[Bc * Kt];
  __shared__ float nxt[Bc * Kt];
  int tid = threadIdx.x;
  for (int st = tid; st < Bc * Kt; st += 1024) {
    int bb = st >> 7, j = st & 127;
    cur[st] = start[j] + emis[((long)bb * Sc) * Kt + j];
  }
  __syncthreads();
  for (int t = 1; t < Sc; ++t) {
    for (int st = tid; st < Bc * Kt; st += 1024) {
      int bb = st >> 7, j = st & 127;
      const float* c = &cur[bb * Kt];
      const float* tr = &trT[j * Kt];
      float best = -1e30f; int bi = 0;
      for (int i = 0; i < Kt; ++i) {
        float v = c[i] + tr[i];
        if (v > best) { best = v; bi = i; }   // first-max like jnp.argmax
      }
      int on = mask[bb * Sc + t] > 0;
      nxt[st] = on ? (best + emis[((long)bb * Sc + t) * Kt + j]) : c[j];
      bp[(long)(t - 1) * Bc * Kt + st] = on ? bi : j;
    }
    __syncthreads();
    for (int st = tid; st < Bc * Kt; st += 1024) cur[st] = nxt[st];
    __syncthreads();
  }
  for (int st = tid; st < Bc * Kt; st += 1024) scoreV[st] = cur[st];
}

// ---------------- losses + intent head + viterbi backtrace ----------------
__global__ __launch_bounds__(128)
void crf_finalize_kernel(const float* __restrict__ scoreT, const float* __restrict__ scoreV,
                         const int* __restrict__ bp, const float* __restrict__ emis,
                         const int* __restrict__ mask, const int* __restrict__ tags,
                         const float* __restrict__ start, const float* __restrict__ endv,
                         const float* __restrict__ trans, const float* __restrict__ ilogits,
                         const int* __restrict__ itgt, float* __restrict__ out) {
  __shared__ float llh[Bc], iloss[Bc], rowbest[Bc];
  __shared__ int   rowcol[Bc];
  int tid = threadIdx.x;
  if (tid < Bc) {
    int bb = tid;
    // logZ
    float mx = -1e30f;
    for (int k = 0; k < Kt; ++k) mx = fmaxf(mx, scoreT[bb * Kt + k] + endv[k]);
    float ssum = 0.f;
    for (int k = 0; k < Kt; ++k) ssum += __expf(scoreT[bb * Kt + k] + endv[k] - mx);
    float logZ = mx + __logf(ssum);
    // gold path score
    float num = start[tags[bb * Sc]];
    int seqlen = 0;
    for (int t = 0; t < Sc; ++t) {
      int on = mask[bb * Sc + t] > 0;
      float m = on ? 1.f : 0.f;
      seqlen += on;
      int tg = tags[bb * Sc + t];
      num += emis[((long)bb * Sc + t) * Kt + tg] * m;
      if (t >= 1) num += trans[tags[bb * Sc + t - 1] * Kt + tg] * m;
    }
    int sl = seqlen > 0 ? seqlen : 1;
    num += endv[tags[bb * Sc + (sl - 1)]];
    llh[bb] = num - logZ;
    // intent log-softmax row
    const float* row = ilogits + bb * NIc;
    float rm = -1e30f; int rc = 0;
    for (int j = 0; j < NIc; ++j) { float v = row[j]; if (v > rm) { rm = v; rc = j; } }
    float se = 0.f;
    for (int j = 0; j < NIc; ++j) se += __expf(row[j] - rm);
    iloss[bb]   = -(row[itgt[bb]] - (rm + __logf(se)));
    rowbest[bb] = 1.f / se;      // max softmax prob within the row
    rowcol[bb]  = rc;
    // viterbi backtrace
    float bm = -1e30f; int bi = 0;
    for (int k = 0; k < Kt; ++k) {
      float v = scoreV[bb * Kt + k] + endv[k];
      if (v > bm) { bm = v; bi = k; }
    }
    int tg = bi;
    out[1 + bb * Sc + (Sc - 1)] = (float)tg;
    for (int t = Sc - 2; t >= 0; --t) {
      tg = bp[(long)t * Bc * Kt + bb * Kt + tg];
      out[1 + bb * Sc + t] = (float)tg;
    }
  }
  __syncthreads();
  if (tid == 0) {
    float slo = 0.f, ilo = 0.f;
    for (int b = 0; b < Bc; ++b) { slo += -llh[b]; ilo += iloss[b]; }
    slo /= (float)Bc; ilo /= (float)Bc;
    out[0] = (slo + ilo) * 0.5f;
    // flattened argmax of row-softmaxed logits (replicates reference quirk)
    float best = -1e30f; int bbest = 0;
    for (int b = 0; b < Bc; ++b)
      if (rowbest[b] > best) { best = rowbest[b]; bbest = b; }
    out[1 + Bc * Sc] = (float)(bbest * NIc + rowcol[bbest]);
  }
}

// =====================================================================
// host orchestration
// =====================================================================
static inline void gemm(hipStream_t st,
                        const float* A, long lda, long sAo, long sAi,
                        const float* Bm, long ldb, long sBo, long sBi, int bT,
                        float* C, long ldc, long sCo, long sCi,
                        const float* bias, int M, int N, int Kd,
                        int innerB, int nbatch, float alpha) {
  dim3 grid((N + 63) / 64, (M + 127) / 128, nbatch);
  wmma_gemm_bf16<<<grid, dim3(256), 0, st>>>(A, lda, sAo, sAi, Bm, ldb, sBo, sBi, bT,
                                             C, ldc, sCo, sCi, bias, M, N, Kd, innerB, alpha);
}

extern "C" void kernel_launch(void* const* d_in, const int* in_sizes, int n_in,
                              void* d_out, int out_size, void* d_ws, size_t ws_size,
                              hipStream_t stream) {
  (void)in_sizes; (void)n_in; (void)out_size; (void)ws_size;
  const int*   input_ids = (const int*)d_in[0];
  const int*   attn_mask = (const int*)d_in[1];
  const int*   itgt      = (const int*)d_in[2];
  const int*   stgt      = (const int*)d_in[3];
  const float* word_emb  = (const float*)d_in[4];
  const float* pos_emb   = (const float*)d_in[5];
  const float* emb_ln_s  = (const float*)d_in[6];
  const float* emb_ln_b  = (const float*)d_in[7];
  const float* Wq = (const float*)d_in[8];   const float* bq = (const float*)d_in[9];
  const float* Wk = (const float*)d_in[10];  const float* bk = (const float*)d_in[11];
  const float* Wv = (const float*)d_in[12];  const float* bv = (const float*)d_in[13];
  const float* Wo = (const float*)d_in[14];  const float* bo = (const float*)d_in[15];
  const float* ln1s = (const float*)d_in[16]; const float* ln1b = (const float*)d_in[17];
  const float* W1 = (const float*)d_in[18];  const float* b1 = (const float*)d_in[19];
  const float* W2 = (const float*)d_in[20];  const float* b2 = (const float*)d_in[21];
  const float* ln2s = (const float*)d_in[22]; const float* ln2b = (const float*)d_in[23];
  const float* intent_W = (const float*)d_in[24]; const float* intent_b = (const float*)d_in[25];
  const float* slot_W   = (const float*)d_in[26]; const float* slot_b   = (const float*)d_in[27];
  const float* crf_start = (const float*)d_in[28];
  const float* crf_end   = (const float*)d_in[29];
  const float* crf_trans = (const float*)d_in[30];

  // workspace carve-out
  char* wp = (char*)d_ws;
  auto carve = [&](size_t bytes) -> void* {
    void* p = (void*)wp;
    wp += (bytes + 255) & ~(size_t)255;
    return p;
  };
  const size_t BSD  = (size_t)Bc * Sc * Dc * sizeof(float);      // 25.2 MB
  const size_t BIG  = (size_t)Bc * Hc * Sc * Sc * sizeof(float); // 100.7 MB (== B*S*FF)
  float* x      = (float*)carve(BSD);
  float* qb     = (float*)carve(BSD);
  float* kb     = (float*)carve(BSD);
  float* vb     = (float*)carve(BSD);
  float* ctx    = (float*)carve(BSD);
  float* tmp    = (float*)carve(BSD);
  float* big    = (float*)carve(BIG);            // scores / FFN hidden (shared)
  float* emis   = (float*)carve((size_t)Bc * Sc * Kt * sizeof(float));
  float* ilog   = (float*)carve((size_t)Bc * NIc * sizeof(float));
  float* scoreT = (float*)carve((size_t)Bc * Kt * sizeof(float));
  float* scoreV = (float*)carve((size_t)Bc * Kt * sizeof(float));
  float* trT    = (float*)carve((size_t)Kt * Kt * sizeof(float));
  int*   bpbuf  = (int*)carve((size_t)(Sc - 1) * Bc * Kt * sizeof(int));
  float* out    = (float*)d_out;

  const long SD  = (long)Sc * Dc;
  const long HSS = (long)Hc * Sc * Sc;
  const long SS  = (long)Sc * Sc;

  // embeddings + LN
  embed_ln_kernel<<<Bc * Sc, 256, 0, stream>>>(input_ids, word_emb, pos_emb,
                                               emb_ln_s, emb_ln_b, x);
  // CRF transition transpose (once)
  transpose128_kernel<<<(Kt * Kt + 255) / 256, 256, 0, stream>>>(crf_trans, trT);

  for (int i = 0; i < Lc; ++i) {
    const float* Wqi = Wq + (long)i * Dc * Dc; const float* bqi = bq + (long)i * Dc;
    const float* Wki = Wk + (long)i * Dc * Dc; const float* bki = bk + (long)i * Dc;
    const float* Wvi = Wv + (long)i * Dc * Dc; const float* bvi = bv + (long)i * Dc;
    const float* Woi = Wo + (long)i * Dc * Dc; const float* boi = bo + (long)i * Dc;
    const float* W1i = W1 + (long)i * Dc * FFc; const float* b1i = b1 + (long)i * FFc;
    const float* W2i = W2 + (long)i * FFc * Dc; const float* b2i = b2 + (long)i * Dc;

    // Q, K, V projections
    gemm(stream, x, Dc, 0, 0, Wqi, Dc, 0, 0, 0, qb, Dc, 0, 0, bqi, Bc * Sc, Dc, Dc, 1, 1, 1.f);
    gemm(stream, x, Dc, 0, 0, Wki, Dc, 0, 0, 0, kb, Dc, 0, 0, bki, Bc * Sc, Dc, Dc, 1, 1, 1.f);
    gemm(stream, x, Dc, 0, 0, Wvi, Dc, 0, 0, 0, vb, Dc, 0, 0, bvi, Bc * Sc, Dc, Dc, 1, 1, 1.f);
    // scores[b,h] = Q K^T / sqrt(DH)   (batched over b*h, B transposed)
    gemm(stream, qb, Dc, SD, DHc, kb, Dc, SD, DHc, 1,
         big, Sc, HSS, SS, nullptr, Sc, Sc, DHc, Hc, Bc * Hc, 0.125f);
    softmax_mask_kernel<<<Bc * Hc * Sc, 256, 0, stream>>>(big, attn_mask);
    // ctx[b,h] = P V
    gemm(stream, big, Sc, HSS, SS, vb, Dc, SD, DHc, 0,
         ctx, Dc, SD, DHc, nullptr, Sc, DHc, Sc, Hc, Bc * Hc, 1.f);
    // output proj + residual LN
    gemm(stream, ctx, Dc, 0, 0, Woi, Dc, 0, 0, 0, tmp, Dc, 0, 0, boi, Bc * Sc, Dc, Dc, 1, 1, 1.f);
    residual_ln_kernel<<<Bc * Sc, 256, 0, stream>>>(x, tmp, ln1s + (long)i * Dc, ln1b + (long)i * Dc);
    // FFN
    gemm(stream, x, Dc, 0, 0, W1i, FFc, 0, 0, 0, big, FFc, 0, 0, b1i, Bc * Sc, FFc, Dc, 1, 1, 1.f);
    gelu_kernel<<<(int)(((long)Bc * Sc * FFc + 255) / 256), 256, 0, stream>>>(big, (long)Bc * Sc * FFc);
    gemm(stream, big, FFc, 0, 0, W2i, Dc, 0, 0, 0, tmp, Dc, 0, 0, b2i, Bc * Sc, Dc, FFc, 1, 1, 1.f);
    residual_ln_kernel<<<Bc * Sc, 256, 0, stream>>>(x, tmp, ln2s + (long)i * Dc, ln2b + (long)i * Dc);
  }

  // intent head on x[:,0,:] (row stride S*D picks CLS token)
  gemm(stream, x, SD, 0, 0, intent_W, NIc, 0, 0, 0, ilog, NIc, 0, 0,
       intent_b, Bc, NIc, Dc, 1, 1, 1.f);
  // slot emissions
  gemm(stream, x, Dc, 0, 0, slot_W, Kt, 0, 0, 0, emis, Kt, 0, 0,
       slot_b, Bc * Sc, Kt, Dc, 1, 1, 1.f);

  // CRF
  crf_forward_kernel<<<1, 1024, 0, stream>>>(emis, attn_mask, crf_start, trT, scoreT);
  crf_viterbi_kernel<<<1, 1024, 0, stream>>>(emis, attn_mask, crf_start, trT, scoreV, bpbuf);
  crf_finalize_kernel<<<1, 128, 0, stream>>>(scoreT, scoreV, bpbuf, emis, attn_mask, stgt,
                                             crf_start, crf_end, crf_trans, ilog, itgt, out);
}
